// TorsionAttention_32667521253874
// MI455X (gfx1250) — matrix operand order repaired and checked by reading, not compiled, not run on hardware
//
#include <hip/hip_runtime.h>
#include <hip/hip_bf16.h>
#include <math.h>

// ---------------- problem constants ----------------
#define DIM   1024
#define LSEQ  1024
#define BSZ   4
#define NH    16
#define HD    64
#define BH    (BSZ * NH)     // 64
#define BLROWS (BSZ * LSEQ)  // 4096

typedef __attribute__((ext_vector_type(16))) __bf16 v16bf;
typedef __attribute__((ext_vector_type(8)))  float  v8f;

#if __has_builtin(__builtin_amdgcn_global_load_async_to_lds_b32)
#define HAVE_ASYNC_LDS 1
#endif

// ---------------- helpers ----------------
__device__ __forceinline__ unsigned short bf16_bits(float f) {
  __bf16 h = (__bf16)f;                       // hardware RNE conversion
  return __builtin_bit_cast(unsigned short, h);
}
__device__ __forceinline__ unsigned pack_bf16x2(float lo, float hi) {
#if __has_builtin(__builtin_amdgcn_cvt_pk_bf16_f32)
  auto r = __builtin_amdgcn_cvt_pk_bf16_f32(lo, hi);   // v_cvt_pk_bf16_f32
  return __builtin_bit_cast(unsigned, r);
#else
  return (unsigned)bf16_bits(lo) | ((unsigned)bf16_bits(hi) << 16);
#endif
}

#ifdef HAVE_ASYNC_LDS
__device__ __forceinline__ void wait_async0() {
#if __has_builtin(__builtin_amdgcn_s_wait_asynccnt)
  __builtin_amdgcn_s_wait_asynccnt(0);
#else
  asm volatile("s_wait_asynccnt 0" ::: "memory");
#endif
}
__device__ __forceinline__ void async_copy_b32(const void* g, void* l) {
  __builtin_amdgcn_global_load_async_to_lds_b32(
      (__attribute__((address_space(1))) int*)g,
      (__attribute__((address_space(3))) int*)l, 0, 0);
}
#endif

// WMMA 16x16x32 bf16 fragment layouts (CDNA5 ISA §7.12.2), offsets in halves
// A (16x32, MxK): lane = m + 16*((k>>3)&1) ; elem = (k&7) + 8*(k>>4)
__device__ __forceinline__ int fragA_off(int m, int k) {
  return ((m + (((k >> 3) & 1) << 4)) << 4) + ((k & 7) | ((k >> 4) << 3));
}
// B (32x16, KxN): lane = n + 16*((k>>3)&1) ; elem = (k&7) + 8*(k>>4)
__device__ __forceinline__ int fragB_off(int n, int k) {
  return ((n + (((k >> 3) & 1) << 4)) << 4) + ((k & 7) | ((k >> 4) << 3));
}

__device__ __forceinline__ v8f wmma_bf16(const unsigned short* a, const unsigned short* b, v8f c) {
  v16bf a16 = *(const v16bf*)a;
  v16bf b16 = *(const v16bf*)b;
  return __builtin_amdgcn_wmma_f32_16x16x32_bf16(false, a16, false, b16, (short)0, c, false, false);
}

// ---------------- kernel 0: torsion bias vector + energy ----------------
__global__ __launch_bounds__(256) void bias_energy_kernel(
    const float* __restrict__ field, const float* __restrict__ strength,
    float* __restrict__ biasbuf, float* __restrict__ energy_out) {
  int l = blockIdx.x * 256 + threadIdx.x;
  const float PI = 3.14159265358979323846f;
  if (l < LSEQ) {
    float pos = (float)l;
    float acc = 0.f;
    #pragma unroll
    for (int i = 0; i < 3; ++i) {
      float s   = strength[i];
      float mod = s * sinf(pos * (float)(i + 1) * PI / (float)LSEQ);
      float ph  = 1.f / (1.f + expf(-field[i * DIM + l]));
      acc += mod * ph;
    }
    biasbuf[l] = acc;
  }
  if (blockIdx.x == 0 && threadIdx.x == 0) {
    float s0 = strength[0], s1 = strength[1], s2 = strength[2];
    energy_out[0] = (s0 * s0 + s1 * s1 + s2 * s2) * (1.f / 3.f);
  }
}

// ---------------- kernel 1: fused QKV projection (f32 in -> bf16 out) ----------------
// y[r,e] = sum_d x[r,d] * W[e,d]  (+ bias[e]); stored as (B,H,L,hd) bf16
__global__ __launch_bounds__(256) void qkv_proj_kernel(
    const float* __restrict__ x,
    const float* __restrict__ Wq, const float* __restrict__ bq,
    const float* __restrict__ Wk, const float* __restrict__ bk,
    const float* __restrict__ Wv, const float* __restrict__ bv,
    unsigned short* __restrict__ qb, unsigned short* __restrict__ kb,
    unsigned short* __restrict__ vb) {
  const int tid  = threadIdx.x;
  const int lane = tid & 31;
  const int wave = tid >> 5;
  const int r0 = blockIdx.y * 128;   // row in (B*L)
  const int c0 = blockIdx.x * 128;   // output feature e

  const float* W; const float* bias; unsigned short* dst;
  if (blockIdx.z == 0)      { W = Wq; bias = bq; dst = qb; }
  else if (blockIdx.z == 1) { W = Wk; bias = bk; dst = kb; }
  else                      { W = Wv; bias = bv; dst = vb; }

  __shared__ __align__(32) unsigned short smA[128 * 64];  // 16 tiles of 16x32
  __shared__ __align__(32) unsigned short smB[128 * 64];

  const int wr = wave & 1;   // 2 row groups of 64
  const int wc = wave >> 1;  // 4 col groups of 32
  v8f acc[4][2];
  #pragma unroll
  for (int i = 0; i < 4; ++i)
    #pragma unroll
    for (int j = 0; j < 2; ++j)
      acc[i][j] = (v8f){0.f, 0.f, 0.f, 0.f, 0.f, 0.f, 0.f, 0.f};

  for (int kk0 = 0; kk0 < DIM; kk0 += 64) {
    // stage A: x[r0..r0+127, kk0..kk0+63] -> bf16 fragments
    #pragma unroll
    for (int it = 0; it < 16; ++it) {
      int i = tid + it * 256;              // 0..4095 (float2 index)
      int row = i >> 5, pr = i & 31, k = pr << 1;
      float2 f = ((const float2*)(x + (size_t)(r0 + row) * DIM + kk0))[pr];
      int off = (((k >> 5) << 3) + (row >> 4)) * 512 + fragA_off(row & 15, k & 31);
      *(unsigned*)(smA + off) = pack_bf16x2(f.x, f.y);
    }
    // stage B: W[c0+n, kk0+k] = B[k,n]
    #pragma unroll
    for (int it = 0; it < 16; ++it) {
      int i = tid + it * 256;
      int n = i >> 5, pr = i & 31, k = pr << 1;
      float2 f = ((const float2*)(W + (size_t)(c0 + n) * DIM + kk0))[pr];
      int off = (((k >> 5) << 3) + (n >> 4)) * 512 + fragB_off(n & 15, k & 31);
      *(unsigned*)(smB + off) = pack_bf16x2(f.x, f.y);
    }
    __syncthreads();
    #pragma unroll
    for (int kc = 0; kc < 2; ++kc) {
      #pragma unroll
      for (int cs = 0; cs < 2; ++cs) {
        const unsigned short* bp = smB + (kc * 8 + wc * 2 + cs) * 512 + lane * 16;
        #pragma unroll
        for (int rs = 0; rs < 4; ++rs) {
          const unsigned short* ap = smA + (kc * 8 + wr * 4 + rs) * 512 + lane * 16;
          acc[rs][cs] = wmma_bf16(ap, bp, acc[rs][cs]);
        }
      }
    }
    __syncthreads();
  }

  // epilogue: C layout m = r + 8*(lane>>4), n = lane&15; reshape to (B,H,L,hd) bf16
  const int nn = lane & 15;
  const int g  = lane >> 4;
  #pragma unroll
  for (int rs = 0; rs < 4; ++rs) {
    #pragma unroll
    for (int cs = 0; cs < 2; ++cs) {
      #pragma unroll
      for (int r = 0; r < 8; ++r) {
        int m   = r + (g << 3);
        int row = r0 + wr * 64 + rs * 16 + m;
        int col = c0 + wc * 32 + cs * 16 + nn;
        float val = acc[rs][cs][r] + bias[col];
        int b = row >> 10, l = row & 1023, h = col >> 6, dh = col & 63;
        size_t idx = (((size_t)(b * NH + h) * LSEQ) + l) * HD + dh;
        dst[idx] = bf16_bits(val);
      }
    }
  }
}

// ---------------- kernel 2: fused scores + softmax ----------------
// One block owns 64 query rows of one (b,h): computes S = QK^T*0.125 + bias into
// 256 KiB of LDS (CDNA5 320KB WGP LDS), does the full row softmax in LDS, and
// writes the normalized attn to HBM exactly once (coalesced float4 stores).
__global__ __launch_bounds__(256) void scores_softmax_kernel(
    const unsigned short* __restrict__ qb, const unsigned short* __restrict__ kb,
    const float* __restrict__ biasbuf, float* __restrict__ attn) {
  const int tid  = threadIdx.x;
  const int lane = tid & 31;
  const int wave = tid >> 5;
  const int r0 = blockIdx.x * 64;    // query rows
  const int z  = blockIdx.y;         // b*NH + h

  __shared__ __align__(16) float sS[64 * 1024];            // 256 KiB scores
  __shared__ __align__(32) unsigned short smA[64 * 64];    // Q fragments (8 KiB)
  __shared__ __align__(32) unsigned short smB[128 * 64];   // K fragments (16 KiB)
  __shared__ float red[256];
  __shared__ float rowstat[64];

  // stage Q rows r0..r0+63 (full K = hd = 64) once
  #pragma unroll
  for (int it = 0; it < 8; ++it) {
    int i = tid + it * 256;            // 0..2047 uints
    int row = i >> 5, pr = i & 31, k = pr << 1;
    const unsigned short* g = qb + ((size_t)z * LSEQ + r0 + row) * HD + k;
    int off = (((k >> 5) << 2) + (row >> 4)) * 512 + fragA_off(row & 15, k & 31);
#ifdef HAVE_ASYNC_LDS
    async_copy_b32(g, smA + off);
#else
    *(unsigned*)(smA + off) = *(const unsigned*)g;
#endif
  }

  const int rt   = wave & 3;   // 4 row tiles of 16
  const int cgrp = wave >> 2;  // 2 col groups of 64

  for (int ct = 0; ct < 8; ++ct) {   // loop over 128-wide key tiles
    const int cb = ct * 128;
    #pragma unroll
    for (int it = 0; it < 16; ++it) {
      int i = tid + it * 256;
      int n = i >> 5, pr = i & 31, k = pr << 1;
      const unsigned short* g = kb + ((size_t)z * LSEQ + cb + n) * HD + k;
      int off = (((k >> 5) << 3) + (n >> 4)) * 512 + fragB_off(n & 15, k & 31);
#ifdef HAVE_ASYNC_LDS
      async_copy_b32(g, smB + off);
#else
      *(unsigned*)(smB + off) = *(const unsigned*)g;
#endif
    }
#ifdef HAVE_ASYNC_LDS
    wait_async0();
#endif
    __syncthreads();

    v8f acc[4];
    #pragma unroll
    for (int cs = 0; cs < 4; ++cs)
      acc[cs] = (v8f){0.f, 0.f, 0.f, 0.f, 0.f, 0.f, 0.f, 0.f};
    #pragma unroll
    for (int kc = 0; kc < 2; ++kc) {
      const unsigned short* ap = smA + (kc * 4 + rt) * 512 + lane * 16;
      #pragma unroll
      for (int cs = 0; cs < 4; ++cs) {
        const unsigned short* bp = smB + (kc * 8 + cgrp * 4 + cs) * 512 + lane * 16;
        acc[cs] = wmma_bf16(ap, bp, acc[cs]);
      }
    }
    const int nn = lane & 15, g = lane >> 4;
    #pragma unroll
    for (int cs = 0; cs < 4; ++cs) {
      int col = cb + cgrp * 64 + cs * 16 + nn;
      float bvc = biasbuf[col];
      #pragma unroll
      for (int r = 0; r < 8; ++r) {
        int row = rt * 16 + r + (g << 3);
        sS[row * 1024 + col] = acc[cs][r] * 0.125f + bvc;   // 1/sqrt(64)
      }
    }
    __syncthreads();
  }

  // ---- row softmax over sS (64 rows x 1024), 4 threads per row ----
  const int rrow = tid >> 2, q4 = tid & 3;
  float* rp = sS + rrow * 1024 + q4 * 256;

  float mx = -INFINITY;
  #pragma unroll 8
  for (int i = 0; i < 64; ++i) {
    float4 v = ((const float4*)rp)[i];
    mx = fmaxf(mx, fmaxf(fmaxf(v.x, v.y), fmaxf(v.z, v.w)));
  }
  red[tid] = mx;
  __syncthreads();
  if (q4 == 0)
    rowstat[rrow] = fmaxf(fmaxf(red[tid], red[tid + 1]), fmaxf(red[tid + 2], red[tid + 3]));
  __syncthreads();

  float m = rowstat[rrow];
  float sum = 0.f;
  #pragma unroll 4
  for (int i = 0; i < 64; ++i) {
    float4 v = ((const float4*)rp)[i];
    v.x = expf(v.x - m); v.y = expf(v.y - m);
    v.z = expf(v.z - m); v.w = expf(v.w - m);
    ((float4*)rp)[i] = v;
    sum += (v.x + v.y) + (v.z + v.w);
  }
  red[tid] = sum;
  __syncthreads();
  if (q4 == 0)
    rowstat[rrow] = (red[tid] + red[tid + 1]) + (red[tid + 2] + red[tid + 3]);
  __syncthreads();

  // ---- normalized write: iteration `it` writes row `it`, fully coalesced ----
  float* attnZ = attn + ((size_t)z * LSEQ + r0) * LSEQ;
  for (int it = 0; it < 64; ++it) {
    float inv = 1.f / rowstat[it];
    float4 v = ((const float4*)(sS + it * 1024))[tid];
    v.x *= inv; v.y *= inv; v.z *= inv; v.w *= inv;
    ((float4*)(attnZ + (size_t)it * LSEQ))[tid] = v;
  }
}

// ---------------- kernel 3: O = attn @ V  (per (b,h)) ----------------
__global__ __launch_bounds__(256) void attnv_kernel(
    const float* __restrict__ attn, const unsigned short* __restrict__ vb,
    float* __restrict__ obuf) {
  const int tid  = threadIdx.x;
  const int lane = tid & 31;
  const int wave = tid >> 5;
  const int r0 = blockIdx.x * 128;
  const int z  = blockIdx.y;

  __shared__ __align__(32) unsigned short smA[128 * 64];  // attn tile (bf16 fragments)
  __shared__ __align__(32) unsigned short smB[64 * 64];   // V tile

  const int wr = wave >> 1;  // 4 row groups of 32
  const int wc = wave & 1;   // 2 col groups of 32
  v8f acc[2][2];
  #pragma unroll
  for (int i = 0; i < 2; ++i)
    #pragma unroll
    for (int j = 0; j < 2; ++j)
      acc[i][j] = (v8f){0.f, 0.f, 0.f, 0.f, 0.f, 0.f, 0.f, 0.f};

  for (int kk0 = 0; kk0 < LSEQ; kk0 += 64) {
    #pragma unroll
    for (int it = 0; it < 16; ++it) {
      int i = tid + it * 256;
      int row = i >> 5, pr = i & 31, k = pr << 1;
      float2 f = ((const float2*)(attn + ((size_t)z * LSEQ + r0 + row) * LSEQ + kk0))[pr];
      int off = (((k >> 5) << 3) + (row >> 4)) * 512 + fragA_off(row & 15, k & 31);
      *(unsigned*)(smA + off) = pack_bf16x2(f.x, f.y);
    }
    #pragma unroll
    for (int it = 0; it < 16; ++it) {
      int i = tid + it * 256;          // 0..4095 halves
      int k = i >> 6, n = i & 63;
      unsigned short h = vb[((size_t)z * LSEQ + kk0 + k) * HD + n];
      int off = (((k >> 5) << 2) + (n >> 4)) * 512 + fragB_off(n & 15, k & 31);
      smB[off] = h;
    }
    __syncthreads();
    #pragma unroll
    for (int kc = 0; kc < 2; ++kc) {
      #pragma unroll
      for (int cs = 0; cs < 2; ++cs) {
        const unsigned short* bp = smB + (kc * 4 + wc * 2 + cs) * 512 + lane * 16;
        #pragma unroll
        for (int rs = 0; rs < 2; ++rs) {
          const unsigned short* ap = smA + (kc * 8 + wr * 2 + rs) * 512 + lane * 16;
          acc[rs][cs] = wmma_bf16(ap, bp, acc[rs][cs]);
        }
      }
    }
    __syncthreads();
  }

  const int nn = lane & 15;
  const int g  = lane >> 4;
  #pragma unroll
  for (int rs = 0; rs < 2; ++rs) {
    #pragma unroll
    for (int cs = 0; cs < 2; ++cs) {
      #pragma unroll
      for (int r = 0; r < 8; ++r) {
        int m   = r + (g << 3);
        int row = r0 + wr * 32 + rs * 16 + m;
        int col = wc * 32 + cs * 16 + nn;
        obuf[((size_t)z * LSEQ + row) * HD + col] = acc[rs][cs][r];
      }
    }
  }
}

// ---------------- kernel 4: out = Z @ Wo^T + bo  (Z gathered from obuf) ----------------
__global__ __launch_bounds__(256) void outproj_kernel(
    const float* __restrict__ obuf, const float* __restrict__ Wo,
    const float* __restrict__ bo, float* __restrict__ out) {
  const int tid  = threadIdx.x;
  const int lane = tid & 31;
  const int wave = tid >> 5;
  const int r0 = blockIdx.y * 128;   // row in (B*L)
  const int c0 = blockIdx.x * 128;   // output feature e

  __shared__ __align__(32) unsigned short smA[128 * 64];
  __shared__ __align__(32) unsigned short smB[128 * 64];

  const int wr = wave & 1, wc = wave >> 1;
  v8f acc[4][2];
  #pragma unroll
  for (int i = 0; i < 4; ++i)
    #pragma unroll
    for (int j = 0; j < 2; ++j)
      acc[i][j] = (v8f){0.f, 0.f, 0.f, 0.f, 0.f, 0.f, 0.f, 0.f};

  for (int kk0 = 0; kk0 < DIM; kk0 += 64) {
    #pragma unroll
    for (int it = 0; it < 16; ++it) {
      int i = tid + it * 256;
      int row = i >> 5, pr = i & 31, k = pr << 1;
      int bl = r0 + row, b = bl >> 10, l = bl & 1023;
      int dg = kk0 + k, h = dg >> 6, dh = dg & 63;
      float2 f = *(const float2*)(obuf + ((size_t)(b * NH + h) * LSEQ + l) * HD + dh);
      int off = (((k >> 5) << 3) + (row >> 4)) * 512 + fragA_off(row & 15, k & 31);
      *(unsigned*)(smA + off) = pack_bf16x2(f.x, f.y);
    }
    #pragma unroll
    for (int it = 0; it < 16; ++it) {
      int i = tid + it * 256;
      int n = i >> 5, pr = i & 31, k = pr << 1;
      float2 f = ((const float2*)(Wo + (size_t)(c0 + n) * DIM + kk0))[pr];
      int off = (((k >> 5) << 3) + (n >> 4)) * 512 + fragB_off(n & 15, k & 31);
      *(unsigned*)(smB + off) = pack_bf16x2(f.x, f.y);
    }
    __syncthreads();
    #pragma unroll
    for (int kc = 0; kc < 2; ++kc) {
      #pragma unroll
      for (int cs = 0; cs < 2; ++cs) {
        const unsigned short* bp = smB + (kc * 8 + wc * 2 + cs) * 512 + lane * 16;
        #pragma unroll
        for (int rs = 0; rs < 4; ++rs) {
          const unsigned short* ap = smA + (kc * 8 + wr * 4 + rs) * 512 + lane * 16;
          acc[rs][cs] = wmma_bf16(ap, bp, acc[rs][cs]);
        }
      }
    }
    __syncthreads();
  }

  const int nn = lane & 15;
  const int g  = lane >> 4;
  #pragma unroll
  for (int rs = 0; rs < 4; ++rs) {
    #pragma unroll
    for (int cs = 0; cs < 2; ++cs) {
      #pragma unroll
      for (int r = 0; r < 8; ++r) {
        int m   = r + (g << 3);
        int row = r0 + wr * 64 + rs * 16 + m;
        int col = c0 + wc * 32 + cs * 16 + nn;
        out[(size_t)row * DIM + col] = acc[rs][cs][r] + bo[col];
      }
    }
  }
}

// ---------------- launcher ----------------
extern "C" void kernel_launch(void* const* d_in, const int* in_sizes, int n_in,
                              void* d_out, int out_size, void* d_ws, size_t ws_size,
                              hipStream_t stream) {
  (void)in_sizes; (void)n_in; (void)out_size; (void)ws_size;

  const float* x      = (const float*)d_in[0];
  const float* Wq     = (const float*)d_in[1];
  const float* bq     = (const float*)d_in[2];
  const float* Wk     = (const float*)d_in[3];
  const float* bk     = (const float*)d_in[4];
  const float* Wv     = (const float*)d_in[5];
  const float* bv     = (const float*)d_in[6];
  const float* Wo     = (const float*)d_in[7];
  const float* bo     = (const float*)d_in[8];
  const float* tfield = (const float*)d_in[9];
  const float* tstr   = (const float*)d_in[10];

  float* out    = (float*)d_out;                       // (4,1024,1024)
  float* attn   = out + (size_t)BSZ * LSEQ * DIM;      // (4,16,1024,1024)
  float* energy = attn + (size_t)BH * LSEQ * LSEQ;     // scalar

  char* ws = (char*)d_ws;
  unsigned short* qb  = (unsigned short*)(ws);                 // 8 MiB bf16
  unsigned short* kb  = (unsigned short*)(ws + (size_t)8  * 1024 * 1024);
  unsigned short* vb  = (unsigned short*)(ws + (size_t)16 * 1024 * 1024);
  float* obuf         = (float*)(ws + (size_t)24 * 1024 * 1024);  // 16 MiB f32
  float* biasbuf      = (float*)(ws + (size_t)40 * 1024 * 1024);  // 4 KiB

  bias_energy_kernel<<<dim3(4), dim3(256), 0, stream>>>(tfield, tstr, biasbuf, energy);
  qkv_proj_kernel<<<dim3(8, 32, 3), dim3(256), 0, stream>>>(x, Wq, bq, Wk, bk, Wv, bv,
                                                            qb, kb, vb);
  scores_softmax_kernel<<<dim3(16, BH), dim3(256), 0, stream>>>(qb, kb, biasbuf, attn);
  attnv_kernel<<<dim3(8, BH), dim3(256), 0, stream>>>(attn, vb, obuf);
  outproj_kernel<<<dim3(8, 32), dim3(256), 0, stream>>>(obuf, Wo, bo, out);
}
